// PedestrianEncoder_3289944949287
// MI455X (gfx1250) — matrix operand order repaired and checked by reading, not compiled, not run on hardware
//
#include <hip/hip_runtime.h>
#include <hip/hip_bf16.h>
#include <math.h>

typedef __attribute__((ext_vector_type(16))) _Float16 v16h;
typedef __attribute__((ext_vector_type(8)))  _Float16 v8h;
typedef __attribute__((ext_vector_type(8)))  float    v8f;
typedef __attribute__((ext_vector_type(4)))  float    v4f;

#define T_ 64
#define B_ 32
#define P_ 32
#define E_ 128
#define H_ 128

// Hardware tanh (v_tanh_f32) on gfx1250; guarded so compilation never breaks.
__device__ __forceinline__ float htanh(float x) {
#if __has_builtin(__builtin_amdgcn_tanhf)
  return __builtin_amdgcn_tanhf(x);
#elif __has_builtin(__builtin_amdgcn_tanh_f32)
  return __builtin_amdgcn_tanh_f32(x);
#else
  return tanhf(x);
#endif
}
// sigmoid via tanh identity -> also uses the hardware TRANS unit
__device__ __forceinline__ float sigm(float x) {
  return 0.5f * htanh(0.5f * x) + 0.5f;
}

__device__ __forceinline__ v16h cat8(v8h lo, v8h hv) {
  return __builtin_shufflevector(lo, hv, 0,1,2,3,4,5,6,7,8,9,10,11,12,13,14,15);
}

__device__ __forceinline__ v8f wmma16(v16h a, v16h b, v8f c) {
  return __builtin_amdgcn_wmma_f32_16x16x32_f16(false, a, false, b, (short)0, c,
                                                false, false);
}

// ---------------------------------------------------------------------------
// Masked LSTM over T=64 steps. One workgroup = 16 rows (all steps), 8 waves;
// wave w owns hidden columns [16w,16w+16). Whh lives in VGPRs as pre-packed
// WMMA B-fragments (weight stationary); h lives in LDS as f16.
// Input projection is algebraically folded: Wc = Wih @ W_inp (4H x DIMX).
// ---------------------------------------------------------------------------
template<int DIMX>
__device__ void lstm_body(
    int tile,
    const float* __restrict__ ch0, const float* __restrict__ ch1,
    const float* __restrict__ ch2, const float* __restrict__ ch3,
    int cstride,
    const int*   __restrict__ maskp,
    const float* __restrict__ Winp, const float* __restrict__ binp,
    const float* __restrict__ Wih,  const float* __restrict__ Whh,
    const float* __restrict__ bih,  const float* __restrict__ bhh,
    _Float16* __restrict__ hout, int* __restrict__ okout, int writeok)
{
  __shared__ __align__(16) _Float16 hbuf[16 * H_];  // 16 rows x 128, row-major
  __shared__ __align__(16) float    xs[16][4];      // per-step inputs (row, chan)
  __shared__ __align__(16) int      ms[16];         // per-step mask

  const int tid  = threadIdx.x;
  const int w    = tid >> 5;
  const int lane = tid & 31;
  const int nn   = lane & 15;
  const int hi   = lane >> 4;
  const int ho8  = hi * 8;

  const int n0  = tile * 16;
  const int bb  = n0 / P_;
  const int p0  = n0 % P_;
  const int col = 16 * w + nn;   // this lane's hidden column

  for (int i = tid; i < 16 * H_; i += 256) hbuf[i] = (_Float16)0.0f;

  // One-time: fold input projection, pack Whh B-fragments.
  // B layout (16-bit 32x16): lane holds column nn; element e -> K = (e<8 ? e : e+8) + 8 hi.
  float Wc[4][4];
  float biasc[4];
  v16h  Bf[4][4];
  for (int g = 0; g < 4; ++g) {
    const int gc = g * H_ + col;               // gate column in [0,512)
    float a0 = 0.f, a1 = 0.f, a2 = 0.f, a3 = 0.f, ab = 0.f;
    for (int e = 0; e < E_; ++e) {
      float wv = Wih[gc * E_ + e];
      a0 += wv * Winp[e * DIMX + 0];
      if (DIMX > 1) a1 += wv * Winp[e * DIMX + 1];
      if (DIMX > 2) a2 += wv * Winp[e * DIMX + 2];
      if (DIMX > 3) a3 += wv * Winp[e * DIMX + 3];
      ab += wv * binp[e];
    }
    Wc[g][0] = a0; Wc[g][1] = a1; Wc[g][2] = a2; Wc[g][3] = a3;
    biasc[g] = ab + bih[gc] + bhh[gc];
    for (int ks = 0; ks < 4; ++ks) {
      v16h bf;
      #pragma unroll
      for (int e = 0; e < 16; ++e)
        bf[e] = (_Float16)Whh[gc * H_ + ks * 32 + (e < 8 ? e : e + 8) + hi * 8];
      Bf[g][ks] = bf;
    }
  }

  float cst[8], hst[8];
  #pragma unroll
  for (int r = 0; r < 8; ++r) { cst[r] = 0.f; hst[r] = 0.f; }
  int cnt = 0;

  __syncthreads();

  for (int t = 0; t < T_; ++t) {
    // stage this step's inputs + mask into LDS; prefetch next step's data
    {
      const int c2 = tid >> 4, i = tid & 15;
      const int base  = (t * B_ + bb) * cstride + p0 + i;
      const int mbase = (t * B_ + bb) * P_ + p0 + i;
      const bool pf = (t + 1) < T_;
      if (c2 == 0) {
        xs[i][0] = ch0[base];
        if (pf) __builtin_prefetch(ch0 + base + B_ * cstride, 0, 0);
      } else if (c2 == 1 && DIMX > 1) {
        xs[i][1] = ch1[base];
        if (pf) __builtin_prefetch(ch1 + base + B_ * cstride, 0, 0);
      } else if (c2 == 2 && DIMX > 2) {
        xs[i][2] = ch2[base];
        if (pf) __builtin_prefetch(ch2 + base + B_ * cstride, 0, 0);
      } else if (c2 == 3 && DIMX > 3) {
        xs[i][3] = ch3[base];
        if (pf) __builtin_prefetch(ch3 + base + B_ * cstride, 0, 0);
      } else if (c2 == DIMX) {
        ms[i] = maskp[mbase];
        if (pf) __builtin_prefetch(maskp + mbase + B_ * P_, 0, 0);
      }
    }

    // gates = h @ Whh.T : 4 gate tiles x 4 K-steps = 16 WMMAs per wave
    v8f acc0, acc1, acc2, acc3;
    #pragma unroll
    for (int r = 0; r < 8; ++r) { acc0[r] = 0.f; acc1[r] = 0.f; acc2[r] = 0.f; acc3[r] = 0.f; }
    #pragma unroll
    for (int ks = 0; ks < 4; ++ks) {
      // A layout (16-bit 16x32): lane row = nn; elems 0..7 -> K=kb..kb+7,
      // elems 8..15 -> K=kb+16..kb+23, kb = 32*ks + 8*hi.
      const _Float16* ap = &hbuf[nn * H_ + ks * 32 + ho8];
      v16h a = cat8(*(const v8h*)ap, *(const v8h*)(ap + 16));
      acc0 = wmma16(a, Bf[0][ks], acc0);
      acc1 = wmma16(a, Bf[1][ks], acc1);
      acc2 = wmma16(a, Bf[2][ks], acc2);
      acc3 = wmma16(a, Bf[3][ks], acc3);
    }

    __syncthreads();  // staging visible; all hbuf reads complete

    // gate nonlinearity + masked (h,c) update, directly in C/D layout
    #pragma unroll
    for (int r = 0; r < 8; ++r) {
      const int row = r + ho8;
      v4f xv = *(const v4f*)&xs[row][0];
      float gi = acc0[r] + biasc[0];
      float gf = acc1[r] + biasc[1];
      float gg = acc2[r] + biasc[2];
      float go = acc3[r] + biasc[3];
      #pragma unroll
      for (int cc = 0; cc < DIMX; ++cc) {
        float x = xv[cc];
        gi += Wc[0][cc] * x; gf += Wc[1][cc] * x;
        gg += Wc[2][cc] * x; go += Wc[3][cc] * x;
      }
      float cn = sigm(gf) * cst[r] + sigm(gi) * htanh(gg);
      float hn = sigm(go) * htanh(cn);
      bool  m  = ms[row] > 0;
      cst[r] = m ? cn : cst[r];
      hst[r] = m ? hn : hst[r];
      hbuf[row * H_ + col] = (_Float16)hst[r];
    }
    if (w == 0 && lane < 16) cnt += (ms[lane] > 0) ? 1 : 0;

    __syncthreads();  // h writes visible; xs reads done before restage
  }

  #pragma unroll
  for (int r = 0; r < 8; ++r)
    hout[(n0 + r + ho8) * H_ + col] = (_Float16)hst[r];
  if (writeok && w == 0 && lane < 16)
    okout[n0 + lane] = (cnt >= 2) ? 1 : 0;
}

// Both independent LSTMs in a single 128-workgroup launch (2x parallelism vs
// back-to-back launches; the branch is blockIdx-uniform so barriers are legal).
__global__ __launch_bounds__(256) void lstm_both_kernel(
    const float* __restrict__ rel,
    const float* __restrict__ vx, const float* __restrict__ vy,
    const float* __restrict__ ax, const float* __restrict__ ay,
    const int*   __restrict__ msk,
    const float* __restrict__ W_sp, const float* __restrict__ b_sp,
    const float* __restrict__ Wih_t, const float* __restrict__ Whh_t,
    const float* __restrict__ bih_t, const float* __restrict__ bhh_t,
    const float* __restrict__ W_st, const float* __restrict__ b_st,
    const float* __restrict__ Wih_s, const float* __restrict__ Whh_s,
    const float* __restrict__ bih_s, const float* __restrict__ bhh_s,
    _Float16* __restrict__ hT, _Float16* __restrict__ hS,
    int* __restrict__ okf)
{
  if (blockIdx.x < 64) {
    // traj-rel layout (T,B,2,P): channel c offset = c*P, (t,b) stride = 2*P
    lstm_body<2>(blockIdx.x, rel, rel + P_, rel, rel, 2 * P_, msk,
                 W_sp, b_sp, Wih_t, Whh_t, bih_t, bhh_t, hT, okf, 1);
  } else {
    // state channels (T,B,1,P): (t,b) stride = P
    lstm_body<4>(blockIdx.x - 64, vx, vy, ax, ay, P_, msk,
                 W_st, b_st, Wih_s, Whh_s, bih_s, bhh_s, hS, okf, 0);
  }
}

// ---------------------------------------------------------------------------
// 16x16 output tile GEMM helper: raw accumulate of A(32xK, f16 LDS) with
// W(j,k) row-major f32 in global.  K = 32*ksteps.
// ---------------------------------------------------------------------------
__device__ __forceinline__ v8f tile_gemm(const _Float16* Al, int As, int mt,
                                         const float* __restrict__ W, int Ws,
                                         int ct, int ksteps, int nn, int hi)
{
  v8f acc;
  #pragma unroll
  for (int r = 0; r < 8; ++r) acc[r] = 0.f;
  const int j = ct * 16 + nn;
  for (int ks = 0; ks < ksteps; ++ks) {
    const _Float16* ap = &Al[(mt * 16 + nn) * As + ks * 32 + hi * 8];
    v16h a = cat8(*(const v8h*)ap, *(const v8h*)(ap + 16));
    v16h b;
    #pragma unroll
    for (int e = 0; e < 16; ++e)
      b[e] = (_Float16)W[j * Ws + ks * 32 + (e < 8 ? e : e + 8) + hi * 8];
    acc = wmma16(a, b, acc);
  }
  return acc;
}

// ---------------------------------------------------------------------------
// Fusion MLP + attention pooling; one workgroup per batch (32 rows).
// ---------------------------------------------------------------------------
__global__ __launch_bounds__(256) void head_kernel(
    const _Float16* __restrict__ hT, const _Float16* __restrict__ hS,
    const int* __restrict__ okf,
    const float* __restrict__ F1, const float* __restrict__ f1,
    const float* __restrict__ F2, const float* __restrict__ f2,
    const float* __restrict__ A1, const float* __restrict__ a1,
    const float* __restrict__ A2, const float* __restrict__ a2,
    float* __restrict__ out)
{
  __shared__ __align__(16) char  regionA[32 * 256 * 2]; // comb -> att_in (f16)
  __shared__ __align__(16) char  regionB[32 * 128 * 4]; // fused1(f16) -> s1(f32)
  __shared__ __align__(16) float fusedf[32 * 128];
  __shared__ float scores[32];
  __shared__ int   okl[32];

  _Float16* comb   = (_Float16*)regionA;
  _Float16* attin  = (_Float16*)regionA;
  _Float16* fused1 = (_Float16*)regionB;
  float*    s1f    = (float*)regionB;

  const int tid = threadIdx.x;
  const int w = tid >> 5, lane = tid & 31, nn = lane & 15, hi = lane >> 4;
  const int ho8 = hi * 8;
  const int bI = blockIdx.x;

  for (int idx = tid; idx < 32 * 256; idx += 256) {
    int p = idx >> 8, k = idx & 255;
    float v = (k < H_) ? (float)hT[(bI * P_ + p) * H_ + k]
                       : (float)hS[(bI * P_ + p) * H_ + (k - H_)];
    comb[idx] = (_Float16)v;
  }
  if (tid < 32) okl[tid] = okf[bI * P_ + tid];
  __syncthreads();

  // fused1 = relu(comb @ F1.T + f1) : 32x128, K=256
  #pragma unroll
  for (int rep = 0; rep < 2; ++rep) {
    int tidx = w + 8 * rep;
    int mt = tidx & 1, ct = tidx >> 1;
    v8f acc = tile_gemm(comb, 256, mt, F1, 256, ct, 8, nn, hi);
    int j = ct * 16 + nn;
    float bv = f1[j];
    #pragma unroll
    for (int r = 0; r < 8; ++r)
      fused1[(mt * 16 + r + ho8) * H_ + j] = (_Float16)fmaxf(acc[r] + bv, 0.f);
  }
  __syncthreads();

  // fused = relu(fused1 @ F2.T + f2), zeroed where !ok : 32x128, K=128
  #pragma unroll
  for (int rep = 0; rep < 2; ++rep) {
    int tidx = w + 8 * rep;
    int mt = tidx & 1, ct = tidx >> 1;
    v8f acc = tile_gemm(fused1, 128, mt, F2, 128, ct, 4, nn, hi);
    int j = ct * 16 + nn;
    float bv = f2[j];
    #pragma unroll
    for (int r = 0; r < 8; ++r) {
      int p = mt * 16 + r + ho8;
      float v = fmaxf(acc[r] + bv, 0.f);
      if (!okl[p]) v = 0.f;
      fusedf[p * H_ + j] = v;
      attin[p * 256 + j] = (_Float16)v;   // regionA: comb is dead now
    }
  }
  __syncthreads();

  // mean over valid rows; broadcast into att_in[:,128:]
  if (tid < H_) {
    float s = 0.f; int nv = 0;
    for (int p = 0; p < P_; ++p)
      if (okl[p]) { s += fusedf[p * H_ + tid]; ++nv; }
    _Float16 mh = (_Float16)(s / fmaxf((float)nv, 1.0f));
    for (int p = 0; p < P_; ++p) attin[p * 256 + H_ + tid] = mh;
  }
  __syncthreads();

  // s1 = relu(att_in @ A1.T + a1) : 32x128, K=256  (overwrites regionB)
  #pragma unroll
  for (int rep = 0; rep < 2; ++rep) {
    int tidx = w + 8 * rep;
    int mt = tidx & 1, ct = tidx >> 1;
    v8f acc = tile_gemm(attin, 256, mt, A1, 256, ct, 8, nn, hi);
    int j = ct * 16 + nn;
    float bv = a1[j];
    #pragma unroll
    for (int r = 0; r < 8; ++r)
      s1f[(mt * 16 + r + ho8) * H_ + j] = fmaxf(acc[r] + bv, 0.f);
  }
  __syncthreads();

  if (tid < P_) {
    float s = 0.f;
    for (int k = 0; k < H_; ++k) s += s1f[tid * H_ + k] * A2[k];
    s = fmaxf(s + a2[0], 0.f);
    scores[tid] = okl[tid] ? s : -1.0e9f;
  }
  __syncthreads();

  if (tid < H_) {
    float mx = -3.0e38f;
    for (int p = 0; p < P_; ++p) mx = fmaxf(mx, scores[p]);
    float se = 0.f, ac = 0.f;
    for (int p = 0; p < P_; ++p) {
      float e = okl[p] ? __expf(scores[p] - mx) : 0.f;
      se += e;
      ac += e * fusedf[p * H_ + tid];
    }
    out[bI * H_ + tid] = ac / fmaxf(se, 1e-9f);
  }
}

extern "C" void kernel_launch(void* const* d_in, const int* in_sizes, int n_in,
                              void* d_out, int out_size, void* d_ws, size_t ws_size,
                              hipStream_t stream)
{
  (void)in_sizes; (void)n_in; (void)out_size; (void)ws_size;

  const float* rel  = (const float*)d_in[1];
  const float* vx   = (const float*)d_in[2];
  const float* vy   = (const float*)d_in[3];
  const float* axp  = (const float*)d_in[4];
  const float* ayp  = (const float*)d_in[5];
  const int*   msk  = (const int*)d_in[6];
  const float* W_sp = (const float*)d_in[7];
  const float* b_sp = (const float*)d_in[8];
  const float* W_st = (const float*)d_in[9];
  const float* b_st = (const float*)d_in[10];
  const float* Wih_t = (const float*)d_in[11];
  const float* Whh_t = (const float*)d_in[12];
  const float* bih_t = (const float*)d_in[13];
  const float* bhh_t = (const float*)d_in[14];
  const float* Wih_s = (const float*)d_in[15];
  const float* Whh_s = (const float*)d_in[16];
  const float* bih_s = (const float*)d_in[17];
  const float* bhh_s = (const float*)d_in[18];
  const float* A1 = (const float*)d_in[19];
  const float* a1 = (const float*)d_in[20];
  const float* A2 = (const float*)d_in[21];
  const float* a2 = (const float*)d_in[22];
  const float* F1 = (const float*)d_in[23];
  const float* f1 = (const float*)d_in[24];
  const float* F2 = (const float*)d_in[25];
  const float* f2 = (const float*)d_in[26];

  char* ws = (char*)d_ws;
  _Float16* hT  = (_Float16*)ws;                         // 256 KB
  _Float16* hS  = (_Float16*)(ws + 1024 * 128 * 2);      // 256 KB
  int*      okf = (int*)(ws + 2 * (1024 * 128 * 2));     //   4 KB

  lstm_both_kernel<<<128, 256, 0, stream>>>(
      rel, vx, vy, axp, ayp, msk,
      W_sp, b_sp, Wih_t, Whh_t, bih_t, bhh_t,
      W_st, b_st, Wih_s, Whh_s, bih_s, bhh_s,
      hT, hS, okf);
  head_kernel<<<32, 256, 0, stream>>>(
      hT, hS, okf, F1, f1, F2, f2, A1, a1, A2, a2, (float*)d_out);
}
